// Lift_45028437131811
// MI455X (gfx1250) — matrix operand order repaired and checked by reading, not compiled, not run on hardware
//
#include <hip/hip_runtime.h>
#include <hip/hip_bf16.h>
#include <math.h>

// ---------------------------------------------------------------------------
// LSS / BEV-pool for MI455X (gfx1250, wave32, WMMA).
//  Phase 1: zero output grid (atomics accumulate into it).
//  Phase 2: per-camera 3x3 inverses (24 threads).
//  Phase 3: GEMM y = W@x + b via v_wmma_f32_16x16x32_f16, fused softmax,
//           depth/feat written channel-major to workspace.
//  Phase 4: frustum geometry + global_atomic_add_f32 scatter into d_out.
// MI455X note: x (34.6MB) + output grid (128MB) are L2-resident (192MB L2),
// so scatter atomics and GEMM re-reads run at L2 speed; simple is right.
// ---------------------------------------------------------------------------

typedef __attribute__((ext_vector_type(16))) _Float16 v16h;
typedef __attribute__((ext_vector_type(8)))  _Float16 v8h;
typedef __attribute__((ext_vector_type(8)))  float    v8f;

#define B_     4
#define N_     6
#define NIMG   24          // B_*N_
#define DCH    41          // depth bins
#define COUT   64          // feature channels
#define OCH    105         // DCH + COUT
#define KDIM   512
#define FH     16
#define FW     44
#define NPIX   704         // FH*FW
#define NXg    128
#define NYg    128
#define NZg    8
#define MTILES 7           // 112 padded output channels / 16
#define NTILES 44          // 704 / 16
#define KCHUNK 128
#define LDA    (KCHUNK + 8)  // halfs per LDS row of W chunk (pad: no bank conflicts)
#define YLD    113           // padded per-pixel channel stride in Y LDS

#define MATS_FLOATS (NIMG * 24)          // per cam: invPR(9) comb(9) tr(3) pt(3)
#define NPTS (NIMG * DCH * NPIX)         // 692736 points
#define OUT_ELEMS (B_ * COUT * NYg * NXg * NZg)   // 33,554,432
#define CSTRIDE (NYg * NXg * NZg)        // 131072: channel stride in output

// ------------------------------ zero output --------------------------------
__global__ void lss_zero(float4* __restrict__ out, int n4) {
  int i = blockIdx.x * blockDim.x + threadIdx.x;
  int stride = gridDim.x * blockDim.x;
  float4 z = make_float4(0.f, 0.f, 0.f, 0.f);
  for (; i < n4; i += stride) out[i] = z;
}

// ------------------------- small matrix helpers ----------------------------
__device__ inline void inv3(const float* a, float* o) {
  float c00 =  (a[4]*a[8] - a[5]*a[7]);
  float c01 = -(a[3]*a[8] - a[5]*a[6]);
  float c02 =  (a[3]*a[7] - a[4]*a[6]);
  float det = a[0]*c00 + a[1]*c01 + a[2]*c02;
  float id  = 1.0f / det;
  o[0] = c00*id;                      o[1] = -(a[1]*a[8]-a[2]*a[7])*id;  o[2] =  (a[1]*a[5]-a[2]*a[4])*id;
  o[3] = c01*id;                      o[4] =  (a[0]*a[8]-a[2]*a[6])*id;  o[5] = -(a[0]*a[5]-a[2]*a[3])*id;
  o[6] = c02*id;                      o[7] = -(a[0]*a[7]-a[1]*a[6])*id;  o[8] =  (a[0]*a[4]-a[1]*a[3])*id;
}
__device__ inline void mm3(const float* a, const float* b, float* o) {
  for (int r = 0; r < 3; ++r)
    for (int c = 0; c < 3; ++c)
      o[r*3+c] = a[r*3+0]*b[0*3+c] + a[r*3+1]*b[1*3+c] + a[r*3+2]*b[2*3+c];
}

// per-camera geometry constants -> ws_mats[bn*24]
__global__ void lss_setup(const float* __restrict__ intrins,
                          const float* __restrict__ imu2c,
                          const float* __restrict__ post_rots,
                          const float* __restrict__ post_trans,
                          float* __restrict__ mats) {
  int t = threadIdx.x;
  if (t >= NIMG) return;
  float K[9], iK[9];
  for (int i = 0; i < 9; ++i) K[i] = intrins[t*9 + i];
  inv3(K, iK);
  const float* M4 = imu2c + t*16;           // 4x4, bottom row [0 0 0 1]
  float A[9], bv[3];
  for (int r = 0; r < 3; ++r) {
    for (int c = 0; c < 3; ++c) A[r*3+c] = M4[r*4+c];
    bv[r] = M4[r*4+3];
  }
  float R[9]; inv3(A, R);                   // rots of c2imu = inv(imu2c)
  float tr[3];
  for (int r = 0; r < 3; ++r) tr[r] = -(R[r*3+0]*bv[0] + R[r*3+1]*bv[1] + R[r*3+2]*bv[2]);
  float C[9]; mm3(R, iK, C);                // combine = rots @ inv(K)
  float PR[9], iPR[9];
  for (int i = 0; i < 9; ++i) PR[i] = post_rots[t*9 + i];
  inv3(PR, iPR);
  float* o = mats + t*24;
  for (int i = 0; i < 9; ++i) o[i]     = iPR[i];
  for (int i = 0; i < 9; ++i) o[9+i]   = C[i];
  for (int i = 0; i < 3; ++i) o[18+i]  = tr[i];
  for (int i = 0; i < 3; ++i) o[21+i]  = post_trans[t*3 + i];
}

// ----------------- GEMM (WMMA f16->f32) + softmax + pack -------------------
// block = 128 threads (4 waves). wave task = (bn, 16-pixel column).
// Each wave: acc[7] v8f covers all 112 (padded) output channels for its
// 16 pixels; K loop in 4 chunks of 128, W chunk staged to LDS as f16.
__global__ void __launch_bounds__(128)
lss_gemm(const float* __restrict__ x,   // [24][512][704]
         const float* __restrict__ W,   // [105][512]
         const float* __restrict__ bias,// [105]
         float* __restrict__ df) {      // [24][105][704] depth(41)+feat(64)
  __shared__ _Float16 Wlds[112 * LDA];       // 30,464 B
  __shared__ float    Ylds[4][16 * YLD];     // 28,928 B

  const int tid  = threadIdx.x;
  const int wv   = tid >> 5;
  const int lane = tid & 31;
  const int n    = lane & 15;      // pixel-in-column / M-row-in-tile
  const int hi   = lane >> 4;      // half-wave selector
  const int task = blockIdx.x * 4 + wv;
  const int bn   = task / NTILES;
  const int ncol = task % NTILES;
  const int pix  = ncol * 16 + n;

  const float* Xp = x + (size_t)bn * KDIM * NPIX;

  v8f acc[MTILES];
  v8f zero = {0.f,0.f,0.f,0.f,0.f,0.f,0.f,0.f};
#pragma unroll
  for (int mt = 0; mt < MTILES; ++mt) acc[mt] = zero;

  const int koffA = hi * 8;    // unused; A is loaded with koffB (contiguous k, as B)
  const int koffB = hi * 16;   // B lane K offset

  for (int kc = 0; kc < KDIM; kc += KCHUNK) {
    // cooperative W chunk -> LDS (f32 -> f16), rows padded m>=105 zeroed
    for (int i = tid; i < 112 * KCHUNK; i += 128) {
      int m  = i >> 7;          // /128
      int kk = i & (KCHUNK - 1);
      float v = (m < OCH) ? W[m * KDIM + kc + kk] : 0.0f;
      Wlds[m * LDA + kk] = (_Float16)v;
    }
    __syncthreads();

#pragma unroll
    for (int kb2 = 0; kb2 < KCHUNK; kb2 += 32) {
      const int kb = kc + kb2;
      // B tile: 32x16 f16; lane holds B[koffB + e][n], e = 0..15
      union { v16h v; _Float16 e[16]; } bu;
#pragma unroll
      for (int e = 0; e < 16; ++e)
        bu.e[e] = (_Float16)Xp[(size_t)(kb + koffB + e) * NPIX + pix];

#pragma unroll
      for (int mt = 0; mt < MTILES; ++mt) {
        // A tile: 16x32 f16; lane m = n; elems 0-15 -> K[koffB + e] (matches B)
        const _Float16* wrow = &Wlds[(mt * 16 + n) * LDA + kb2];
        union { v16h v; v8h h[2]; } au;
        au.h[0] = *(const v8h*)(wrow + koffB);
        au.h[1] = *(const v8h*)(wrow + koffB + 8);
        acc[mt] = __builtin_amdgcn_wmma_f32_16x16x32_f16(
            false, au.v, false, bu.v, (short)0, acc[mt], false, false);
      }
    }
    __syncthreads();   // before next chunk overwrites Wlds
  }

  // spill Y(+bias) to LDS: lane holds D rows r+8*hi of each tile, col n
  float* Y = &Ylds[wv][0];
#pragma unroll
  for (int mt = 0; mt < MTILES; ++mt) {
#pragma unroll
    for (int r = 0; r < 8; ++r) {
      int o = mt * 16 + hi * 8 + r;
      float bb = (o < OCH) ? bias[o] : 0.0f;
      Y[n * YLD + o] = acc[mt][r] + bb;
    }
  }
  __syncthreads();

  float* dfp = df + ((size_t)bn * OCH) * NPIX + pix;
  if (hi == 0) {
    // softmax over 41 depth logits of pixel n
    float mx = -1e30f;
#pragma unroll
    for (int o = 0; o < DCH; ++o) mx = fmaxf(mx, Y[n * YLD + o]);
    float s = 0.f;
#pragma unroll
    for (int o = 0; o < DCH; ++o) s += __expf(Y[n * YLD + o] - mx);
    float inv = 1.0f / s;
#pragma unroll
    for (int o = 0; o < DCH; ++o)
      dfp[(size_t)o * NPIX] = __expf(Y[n * YLD + o] - mx) * inv;
  } else {
    // feature channels 41..104 straight through
#pragma unroll
    for (int c = 0; c < COUT; ++c)
      dfp[(size_t)(DCH + c) * NPIX] = Y[n * YLD + DCH + c];
  }
}

// ---------------------- geometry + atomic scatter --------------------------
__global__ void __launch_bounds__(256)
lss_scatter(const float* __restrict__ mats,
            const float* __restrict__ df,
            float* __restrict__ out) {
  int tid = blockIdx.x * 256 + threadIdx.x;
  if (tid >= NPTS) return;
  int p  = tid % NPIX;
  int t2 = tid / NPIX;
  int d  = t2 % DCH;
  int bn = t2 / DCH;
  int b  = bn / N_;
  int w  = p % FW;
  int h  = p / FW;

  const float* M = mats + bn * 24;
  // frustum point minus post_trans
  float fx = (float)w * (703.0f / 43.0f) - M[21];
  float fy = (float)h * 17.0f            - M[22];
  float fz = (4.0f + (float)d)           - M[23];
  // inv(post_rots) @ p
  float p1x = M[0]*fx + M[1]*fy + M[2]*fz;
  float p1y = M[3]*fx + M[4]*fy + M[5]*fz;
  float p1z = M[6]*fx + M[7]*fy + M[8]*fz;
  // un-project: (u*z, v*z, z)
  float qx = p1x * p1z, qy = p1y * p1z, qz = p1z;
  // combine @ q + trans
  float gx = M[9]*qx  + M[10]*qy + M[11]*qz + M[18];
  float gy = M[12]*qx + M[13]*qy + M[14]*qz + M[19];
  float gz = M[15]*qx + M[16]*qy + M[17]*qz + M[20];
  // voxelize: (g - (BX - DX/2)) / DX, truncate toward zero (matches astype)
  int ix = (int)((gx + 32.25f) * 2.0f);
  int iy = (int)((gy + 32.25f) * 2.0f);
  int iz = (int)( gz + 5.0f);
  if (ix < 0 || ix >= NXg || iy < 0 || iy >= NYg || iz < 0 || iz >= NZg) return;

  // out layout (B, C, NY, NX, NZ)
  size_t base = (size_t)b * COUT * CSTRIDE + ((size_t)iy * NXg + ix) * NZg + iz;
  const float  dval = df[((size_t)bn * OCH + d) * NPIX + p];
  const float* fp   = df + ((size_t)bn * OCH + DCH) * NPIX + p;
#pragma unroll 8
  for (int c = 0; c < COUT; ++c)
    atomicAdd(out + base + (size_t)c * CSTRIDE, dval * fp[(size_t)c * NPIX]);
}

// ------------------------------- launcher ----------------------------------
extern "C" void kernel_launch(void* const* d_in, const int* in_sizes, int n_in,
                              void* d_out, int out_size, void* d_ws, size_t ws_size,
                              hipStream_t stream) {
  (void)in_sizes; (void)n_in; (void)ws_size;
  const float* x          = (const float*)d_in[0];
  const float* intrins    = (const float*)d_in[1];
  const float* imu2c      = (const float*)d_in[2];
  const float* post_rots  = (const float*)d_in[3];
  const float* post_trans = (const float*)d_in[4];
  // d_in[5] undistorts: all-zero, unused by reference math path
  const float* W          = (const float*)d_in[6];
  const float* bias       = (const float*)d_in[7];
  float* out = (float*)d_out;

  float* ws_mats = (float*)d_ws;                 // 576 floats
  float* ws_df   = ws_mats + MATS_FLOATS;        // 24*105*704 floats (~7.1MB)

  lss_zero<<<8192, 256, 0, stream>>>((float4*)out, out_size / 4);
  lss_setup<<<1, 32, 0, stream>>>(intrins, imu2c, post_rots, post_trans, ws_mats);
  lss_gemm<<<(NIMG * NTILES) / 4, 128, 0, stream>>>(x, W, bias, ws_df);
  lss_scatter<<<(NPTS + 255) / 256, 256, 0, stream>>>(ws_mats, ws_df, out);
}
